// T5Attention_3736621547833
// MI455X (gfx1250) — matrix operand order repaired and checked by reading, not compiled, not run on hardware
//
#include <hip/hip_runtime.h>
#include <hip/hip_bf16.h>

// ---------------- problem dims (fixed by the reference) ----------------
constexpr int B_  = 2;
constexpr int T_  = 2048;
constexpr int D_  = 1024;
constexpr int DA_ = 1024;
constexpr int NH_ = 16;
constexpr int HS_ = 64;
#define NEG_MIN_F (-3.4028234663852886e38f)

typedef __attribute__((ext_vector_type(16))) __bf16 v16bf;
typedef __attribute__((ext_vector_type(8)))  float  v8f;

union Frag { v16bf v; uint4 q[2]; };

__device__ __forceinline__ unsigned short f2bf(float f) {
  union { float f; unsigned u; } x; x.f = f;
  unsigned r = x.u + 0x7FFFu + ((x.u >> 16) & 1u);   // round-to-nearest-even
  return (unsigned short)(r >> 16);
}

__device__ __forceinline__ v8f wmma_bf16(v16bf a, v16bf b, v8f c) {
  // D = A(16x32 bf16) * B(32x16 bf16) + C(16x16 f32)
  return __builtin_amdgcn_wmma_f32_16x16x32_bf16(false, a, false, b, (short)0, c,
                                                 false, false);
}

// ---- CDNA5 async global->LDS (ASYNCcnt-tracked), 16B per lane ----
__device__ __forceinline__ void async_g2l_b128(unsigned lds_byte_addr,
                                               const void* gptr) {
  unsigned long long ga = (unsigned long long)(uintptr_t)gptr;
  asm volatile("global_load_async_to_lds_b128 %0, %1, off"
               :: "v"(lds_byte_addr), "v"(ga) : "memory");
}
__device__ __forceinline__ void wait_async_le16() {
  asm volatile("s_wait_asynccnt 0x10" ::: "memory");
}
__device__ __forceinline__ void wait_async_0() {
  asm volatile("s_wait_asynccnt 0x0" ::: "memory");
}

// A-side 16x32 bf16 fragment. ISA layout: lane<16 holds K 0..7 / 16..23 of row
// (lane&15); lane>=16 holds K 8..15 / 24..31.
__device__ __forceinline__ v16bf load_fragA(const unsigned short* p, int stride,
                                            int rowBase, int kOff) {
  const int lane = threadIdx.x & 31;
  const unsigned short* base =
      p + (size_t)(rowBase + (lane & 15)) * stride + kOff + ((lane < 16) ? 0 : 8);
  Frag f;
  f.q[0] = *(const uint4*)(base);        // K k0..k0+7
  f.q[1] = *(const uint4*)(base + 16);   // K k0+16..k0+23
  return f.v;
}

// B-side 32x16 bf16 fragment, sourced as 16 "n" rows each with 32 contiguous k.
// ISA layout: lanes 0-15 hold K 0..7 / 16..23 of col (lane&15); lanes 16-31 hold K 8..15 / 24..31.
__device__ __forceinline__ v16bf load_fragB(const unsigned short* p, int stride,
                                            int rowBase, int kOff) {
  const int lane = threadIdx.x & 31;
  const unsigned short* base =
      p + (size_t)(rowBase + (lane & 15)) * stride + kOff + ((lane < 16) ? 0 : 8);
  Frag f;
  f.q[0] = *(const uint4*)(base);        // K k0..k0+7
  f.q[1] = *(const uint4*)(base + 16);   // K k0+16..k0+23
  return f.v;
}

// =====================================================================
// Kernel 1: fused QKV projection.  y = x @ W^T, rounded to bf16.
// Q,K stored [b,h,t,c]; V stored transposed [b,h,c,t] so the PV GEMM
// reads k-contiguous rows.  64x64 tile per 256-thread block (8 waves,
// each wave = 16x32 of output = 2 WMMA accumulators).
// =====================================================================
__global__ __launch_bounds__(256) void qkv_proj_kernel(
    const float* __restrict__ x,  const float* __restrict__ Wq,
    const float* __restrict__ Wk, const float* __restrict__ Wv,
    unsigned short* __restrict__ qws, unsigned short* __restrict__ kws,
    unsigned short* __restrict__ vtws) {
  __shared__ __align__(16) unsigned short smA[64 * 32];
  __shared__ __align__(16) unsigned short smB[64 * 32];

  const int rowBase = blockIdx.x * 64;   // over M = B*T
  const int colBase = blockIdx.y * 64;   // over N = DA
  const int z = blockIdx.z;              // 0=Q 1=K 2=V
  const float* W = (z == 0) ? Wq : (z == 1) ? Wk : Wv;

  const int tid  = threadIdx.x;
  const int lane = tid & 31;
  const int wave = tid >> 5;
  const int rowBlk = wave >> 1;          // 0..3 -> 16-row slice
  const int colBlk = wave & 1;           // 0..1 -> 32-col slice

  const int srow = tid >> 2;             // staging: 8 elems/thread
  const int scol = (tid & 3) * 8;

  v8f acc0 = {0.f,0.f,0.f,0.f,0.f,0.f,0.f,0.f};
  v8f acc1 = {0.f,0.f,0.f,0.f,0.f,0.f,0.f,0.f};

  for (int kk = 0; kk < D_; kk += 32) {
    {  // stage x tile (64x32 f32 -> bf16)
      const float4* p = (const float4*)(x + (size_t)(rowBase + srow) * D_ + kk + scol);
      float4 f0 = p[0], f1 = p[1];
      unsigned short t8[8] = {f2bf(f0.x), f2bf(f0.y), f2bf(f0.z), f2bf(f0.w),
                              f2bf(f1.x), f2bf(f1.y), f2bf(f1.z), f2bf(f1.w)};
      *(uint4*)&smA[srow * 32 + scol] = *(uint4*)t8;
    }
    {  // stage W tile (64 output-cols x 32 k)
      const float4* p = (const float4*)(W + (size_t)(colBase + srow) * D_ + kk + scol);
      float4 f0 = p[0], f1 = p[1];
      unsigned short t8[8] = {f2bf(f0.x), f2bf(f0.y), f2bf(f0.z), f2bf(f0.w),
                              f2bf(f1.x), f2bf(f1.y), f2bf(f1.z), f2bf(f1.w)};
      *(uint4*)&smB[srow * 32 + scol] = *(uint4*)t8;
    }
    __syncthreads();
    v16bf a  = load_fragA(smA, 32, rowBlk * 16, 0);
    v16bf b0 = load_fragB(smB, 32, colBlk * 32, 0);
    v16bf b1 = load_fragB(smB, 32, colBlk * 32 + 16, 0);
    acc0 = wmma_bf16(a, b0, acc0);
    acc1 = wmma_bf16(a, b1, acc1);
    __syncthreads();
  }

  // C layout: row = p + 8*(lane>=16), col = lane&15.  z-branch hoisted.
  const int hiRow = (lane >= 16) ? 8 : 0;
  const int cl = lane & 15;
  if (z == 2) {
#pragma unroll
    for (int p = 0; p < 8; ++p) {
      const int m = rowBase + rowBlk * 16 + p + hiRow;
      const int b = m >> 11;             // / T_
      const int t = m & (T_ - 1);
#pragma unroll
      for (int nb = 0; nb < 2; ++nb) {
        const int gc = colBase + colBlk * 32 + nb * 16 + cl;
        const int h = gc >> 6, c = gc & 63;
        vtws[(((size_t)(b * NH_ + h)) * HS_ + c) * T_ + t] =
            f2bf(nb == 0 ? acc0[p] : acc1[p]);               // transposed V
      }
    }
  } else {
    unsigned short* __restrict__ ws = (z == 0) ? qws : kws;
#pragma unroll
    for (int p = 0; p < 8; ++p) {
      const int m = rowBase + rowBlk * 16 + p + hiRow;
      const int b = m >> 11;
      const int t = m & (T_ - 1);
#pragma unroll
      for (int nb = 0; nb < 2; ++nb) {
        const int gc = colBase + colBlk * 32 + nb * 16 + cl;
        const int h = gc >> 6, c = gc & 63;
        ws[(((size_t)(b * NH_ + h)) * T_ + t) * HS_ + c] =
            f2bf(nb == 0 ? acc0[p] : acc1[p]);
      }
    }
  }
}

// =====================================================================
// Kernel 2: flash attention.  One wave per (b,h, 16-row Q block).
// K/V tiles double-buffered in LDS via CDNA5 async global->LDS b128
// (ASYNCcnt), next tile prefetched while current one is consumed.
// S = Q K^T (4 WMMA), bias+mask add (dominant 512MB HBM stream),
// online softmax in the replicated C-register layout, P re-swizzled
// through 1KB LDS, O += P V (4 WMMA).
// =====================================================================
__global__ __launch_bounds__(32) void attn_kernel(
    const unsigned short* __restrict__ qws, const unsigned short* __restrict__ kws,
    const unsigned short* __restrict__ vtws, const int* __restrict__ mask,
    const float* __restrict__ pos_bias, unsigned short* __restrict__ aows) {
  __shared__ __align__(16) unsigned short sK[2][32 * 64];  // keys x HS
  __shared__ __align__(16) unsigned short sV[2][64 * 32];  // c x keys
  __shared__ __align__(16) unsigned short sP[16 * 32];

  const int qBase = blockIdx.x * 16;     // query rows
  const int bh = blockIdx.y;             // b*NH + h
  const int b = bh >> 4;
  const int h = bh & (NH_ - 1);
  const int lane = threadIdx.x & 31;
  const int cl = lane & 15;
  const int hiRow = (lane >= 16) ? 8 : 0;

  const unsigned short* qb = qws + (size_t)bh * T_ * HS_;
  const unsigned short* kb = kws + (size_t)bh * T_ * HS_;
  const unsigned short* vb = vtws + (size_t)bh * HS_ * T_;
  const float* biasb = pos_bias + (size_t)h * T_ * T_;
  const int* maskb = mask + b * T_;

  // async-stage K(32x64) and V(64x32) tiles for keys [j, j+32): 16 b128 issues
  auto stage = [&](int buf, int j) {
#pragma unroll
    for (int s = 0; s < 8; ++s) {        // K: 32 rows * 128B = 8 chunks/row
      const int chunk = s * 32 + lane;   // 0..255
      const int row = chunk >> 3;        // key row 0..31
      const int kc = (chunk & 7) * 8;    // element offset in row
      async_g2l_b128((unsigned)(uintptr_t)&sK[buf][row * 64 + kc],
                     kb + (size_t)(j + row) * HS_ + kc);
    }
#pragma unroll
    for (int s = 0; s < 8; ++s) {        // V: 64 rows * 64B = 4 chunks/row
      const int chunk = s * 32 + lane;   // 0..255
      const int row = chunk >> 2;        // c row 0..63
      const int kc = (chunk & 3) * 8;
      async_g2l_b128((unsigned)(uintptr_t)&sV[buf][row * 32 + kc],
                     vb + (size_t)row * T_ + j + kc);
    }
  };

  // Q block held in registers for the whole pass (16x64 = two A fragments)
  const v16bf qf0 = load_fragA(qb, HS_, qBase, 0);
  const v16bf qf1 = load_fragA(qb, HS_, qBase, 32);

  v8f o0 = {0.f,0.f,0.f,0.f,0.f,0.f,0.f,0.f};
  v8f o1 = o0, o2 = o0, o3 = o0;
  float mrun[8], lrun[8];
#pragma unroll
  for (int p = 0; p < 8; ++p) { mrun[p] = -3.0e38f; lrun[p] = 0.f; }

  stage(0, 0);

  for (int j = 0; j < T_; j += 32) {
    const int buf = (j >> 5) & 1;
    if (j + 32 < T_) {
      stage(buf ^ 1, j + 32);   // prefetch next tile (16 more async ops)
      wait_async_le16();        // previous tile complete (loads finish in order)
    } else {
      wait_async_0();
    }

    // ---- S = Q K^T for 32 keys (two 16x16 C tiles) ----
    v16bf k00 = load_fragB(sK[buf], 64, 0, 0);
    v16bf k01 = load_fragB(sK[buf], 64, 0, 32);
    v16bf k10 = load_fragB(sK[buf], 64, 16, 0);
    v16bf k11 = load_fragB(sK[buf], 64, 16, 32);
    v8f s0 = {0.f,0.f,0.f,0.f,0.f,0.f,0.f,0.f};
    v8f s1 = s0;
    s0 = wmma_bf16(qf0, k00, s0);  s0 = wmma_bf16(qf1, k01, s0);
    s1 = wmma_bf16(qf0, k10, s1);  s1 = wmma_bf16(qf1, k11, s1);

    // ---- additive bias with mask fill (dominant HBM stream) ----
    const int col0 = j + cl, col1 = j + 16 + cl;
    const int mk0 = maskb[col0];
    const int mk1 = maskb[col1];
    if (j + 32 < T_)  // prefetch next bias tile -> global_prefetch_b8
      __builtin_prefetch(biasb + (size_t)(qBase + hiRow) * T_ + j + 32 + cl, 0, 0);
#pragma unroll
    for (int p = 0; p < 8; ++p) {
      const float* r = biasb + (size_t)(qBase + hiRow + p) * T_;
      s0[p] += (mk0 == 0) ? NEG_MIN_F : r[col0];
      s1[p] += (mk1 == 0) ? NEG_MIN_F : r[col1];
    }

    // ---- online softmax; stats live in the replicated C layout ----
    float alpha[8];
#pragma unroll
    for (int p = 0; p < 8; ++p) {
      float v = fmaxf(s0[p], s1[p]);
      v = fmaxf(v, __shfl_xor(v, 8, 32));
      v = fmaxf(v, __shfl_xor(v, 4, 32));
      v = fmaxf(v, __shfl_xor(v, 2, 32));
      v = fmaxf(v, __shfl_xor(v, 1, 32));
      const float mn = fmaxf(mrun[p], v);
      alpha[p] = __expf(mrun[p] - mn);
      mrun[p] = mn;
    }
#pragma unroll
    for (int p = 0; p < 8; ++p) {
      const float p0 = __expf(s0[p] - mrun[p]);
      const float p1 = __expf(s1[p] - mrun[p]);
      float rs = p0 + p1;
      rs += __shfl_xor(rs, 8, 32);
      rs += __shfl_xor(rs, 4, 32);
      rs += __shfl_xor(rs, 2, 32);
      rs += __shfl_xor(rs, 1, 32);
      lrun[p] = lrun[p] * alpha[p] + rs;
      const int row = p + hiRow;
      sP[row * 32 + cl] = f2bf(p0);        // C layout -> LDS re-swizzle
      sP[row * 32 + 16 + cl] = f2bf(p1);
      o0[p] *= alpha[p];  o1[p] *= alpha[p];
      o2[p] *= alpha[p];  o3[p] *= alpha[p];
    }

    // ---- O += P V  (P from LDS as A-frag, V rows k-contiguous in LDS) ----
    v16bf pf = load_fragA(sP, 32, 0, 0);
    v16bf vf0 = load_fragB(sV[buf], 32, 0, 0);
    v16bf vf1 = load_fragB(sV[buf], 32, 16, 0);
    v16bf vf2 = load_fragB(sV[buf], 32, 32, 0);
    v16bf vf3 = load_fragB(sV[buf], 32, 48, 0);
    o0 = wmma_bf16(pf, vf0, o0);
    o1 = wmma_bf16(pf, vf1, o1);
    o2 = wmma_bf16(pf, vf2, o2);
    o3 = wmma_bf16(pf, vf3, o3);
  }

  // ---- finalize: divide by row sums, emit bf16 attention output ----
#pragma unroll
  for (int p = 0; p < 8; ++p) {
    const float inv = 1.0f / lrun[p];
    const int qi = qBase + p + hiRow;
    const size_t ro = ((size_t)b * T_ + qi) * DA_ + h * HS_;
    aows[ro + 0 * 16 + cl] = f2bf(o0[p] * inv);
    aows[ro + 1 * 16 + cl] = f2bf(o1[p] * inv);
    aows[ro + 2 * 16 + cl] = f2bf(o2[p] * inv);
    aows[ro + 3 * 16 + cl] = f2bf(o3[p] * inv);
  }
}

// =====================================================================
// Kernel 3: output projection.  out = attn_out @ Wo^T, f32 result.
// Same 64x64 tiling as kernel 1; A is already bf16 in workspace.
// =====================================================================
__global__ __launch_bounds__(256) void out_proj_kernel(
    const unsigned short* __restrict__ ao, const float* __restrict__ Wo,
    float* __restrict__ out) {
  __shared__ __align__(16) unsigned short smA[64 * 32];
  __shared__ __align__(16) unsigned short smB[64 * 32];

  const int rowBase = blockIdx.x * 64;   // over M = B*T
  const int colBase = blockIdx.y * 64;   // over N = D
  const int tid = threadIdx.x;
  const int lane = tid & 31;
  const int wave = tid >> 5;
  const int rowBlk = wave >> 1, colBlk = wave & 1;
  const int srow = tid >> 2;
  const int scol = (tid & 3) * 8;

  v8f acc0 = {0.f,0.f,0.f,0.f,0.f,0.f,0.f,0.f};
  v8f acc1 = acc0;

  for (int kk = 0; kk < DA_; kk += 32) {
    // stage A (already bf16): straight 16B copies
    *(uint4*)&smA[srow * 32 + scol] =
        *(const uint4*)(ao + (size_t)(rowBase + srow) * DA_ + kk + scol);
    {  // stage Wo tile with f32 -> bf16
      const float4* p = (const float4*)(Wo + (size_t)(colBase + srow) * DA_ + kk + scol);
      float4 f0 = p[0], f1 = p[1];
      unsigned short t8[8] = {f2bf(f0.x), f2bf(f0.y), f2bf(f0.z), f2bf(f0.w),
                              f2bf(f1.x), f2bf(f1.y), f2bf(f1.z), f2bf(f1.w)};
      *(uint4*)&smB[srow * 32 + scol] = *(uint4*)t8;
    }
    __syncthreads();
    v16bf a  = load_fragA(smA, 32, rowBlk * 16, 0);
    v16bf b0 = load_fragB(smB, 32, colBlk * 32, 0);
    v16bf b1 = load_fragB(smB, 32, colBlk * 32 + 16, 0);
    acc0 = wmma_bf16(a, b0, acc0);
    acc1 = wmma_bf16(a, b1, acc1);
    __syncthreads();
  }

  const int hiRow = (lane >= 16) ? 8 : 0;
  const int cl = lane & 15;
#pragma unroll
  for (int p = 0; p < 8; ++p) {
    const int m = rowBase + rowBlk * 16 + p + hiRow;
#pragma unroll
    for (int nb = 0; nb < 2; ++nb) {
      const int gc = colBase + colBlk * 32 + nb * 16 + cl;
      out[(size_t)m * D_ + gc] = (nb == 0) ? acc0[p] : acc1[p];
    }
  }
}

// =====================================================================
extern "C" void kernel_launch(void* const* d_in, const int* in_sizes, int n_in,
                              void* d_out, int out_size, void* d_ws, size_t ws_size,
                              hipStream_t stream) {
  const float* x        = (const float*)d_in[0];
  const int*   mask     = (const int*)d_in[1];
  const float* pos_bias = (const float*)d_in[2];
  const float* Wq       = (const float*)d_in[3];
  const float* Wk       = (const float*)d_in[4];
  const float* Wv       = (const float*)d_in[5];
  const float* Wo       = (const float*)d_in[6];
  float* out = (float*)d_out;

  // workspace: 4 bf16 buffers of B*NH*T*HS (= B*T*DA) elements each = 32 MiB
  const size_t seg = (size_t)B_ * NH_ * T_ * HS_;
  unsigned short* qws  = (unsigned short*)d_ws;
  unsigned short* kws  = qws + seg;
  unsigned short* vtws = kws + seg;
  unsigned short* aows = vtws + seg;

  dim3 g1((B_ * T_) / 64, DA_ / 64, 3);
  qkv_proj_kernel<<<g1, 256, 0, stream>>>(x, Wq, Wk, Wv, qws, kws, vtws);

  dim3 g2(T_ / 16, B_ * NH_);
  attn_kernel<<<g2, 32, 0, stream>>>(qws, kws, vtws, mask, pos_bias, aows);

  dim3 g3((B_ * T_) / 64, D_ / 64);
  out_proj_kernel<<<g3, 256, 0, stream>>>(aows, Wo, out);
}
